// OutputLayerProbs_72748156060306
// MI455X (gfx1250) — matrix operand order repaired and checked by reading, not compiled, not run on hardware
//
#include <hip/hip_runtime.h>
#include <hip/hip_bf16.h>
#include <math.h>

typedef __attribute__((ext_vector_type(16))) __bf16 v16bf;
typedef __attribute__((ext_vector_type(8)))  float  v8f;

#define BB 64
#define NN 256
#define DD 512
#define VV 50257
#define KK 100
#define KPAD 112      // 7 tiles of 16 noise columns
#define NTILES 7

__device__ __forceinline__ float softplus_f(float x) {
    // log(1 + e^x), numerically stable
    float ax = fabsf(x);
    return fmaxf(x, 0.0f) + log1pf(__expf(-ax));
}

__global__ void zero_out_kernel(float* __restrict__ out) {
    out[threadIdx.x] = 0.0f;
}

// Convert noise embedding rows to bf16 (padded with zero rows to KPAD) and
// precompute c_n[k] = bias[k] - logV - logprob_noise[k] - logK  (0 for pads).
__global__ void prep_noise_kernel(const float* __restrict__ emb_weight,
                                  const float* __restrict__ emb_bias,
                                  const float* __restrict__ logprob_noise,
                                  const int*   __restrict__ noise_samples,
                                  unsigned short* __restrict__ noise_bf16,
                                  float* __restrict__ c_n) {
    const int k   = blockIdx.x;     // 0..KPAD-1
    const int tid = threadIdx.x;    // 128 threads, 4 floats each
    uint2 packed = make_uint2(0u, 0u);
    if (k < KK) {
        const int row = noise_samples[k];
        float4 v = ((const float4*)(emb_weight + (size_t)row * DD))[tid];
        unsigned int h0 = __builtin_bit_cast(unsigned short, (__bf16)v.x);
        unsigned int h1 = __builtin_bit_cast(unsigned short, (__bf16)v.y);
        unsigned int h2 = __builtin_bit_cast(unsigned short, (__bf16)v.z);
        unsigned int h3 = __builtin_bit_cast(unsigned short, (__bf16)v.w);
        packed = make_uint2(h0 | (h1 << 16), h2 | (h3 << 16));
        if (tid == 0) {
            const float NORM = logf((float)VV);
            const float LOGK = logf((float)KK);
            c_n[k] = emb_bias[row] - NORM - logprob_noise[row] - LOGK;
        }
    } else if (tid == 0) {
        c_n[k] = 0.0f;
    }
    ((uint2*)(noise_bf16 + (size_t)k * DD))[tid] = packed;
}

// One wave handles a tile of 16 consecutive positions (all in the same batch
// row b). WMMA over D in chunks of 32: 7 noise B-tiles + 1 target B-tile
// (diagonal = per-position target dot).
__global__ __launch_bounds__(128)
void nce_main_kernel(const float* __restrict__ input,
                     const float* __restrict__ emb_weight,
                     const float* __restrict__ emb_bias,
                     const float* __restrict__ logprob_noise,
                     const int*   __restrict__ target,
                     const unsigned short* __restrict__ noise_bf16,
                     const float* __restrict__ c_n,
                     float* __restrict__ out) {
    const int lane = threadIdx.x & 31;
    const int wave = threadIdx.x >> 5;
    const int tile = blockIdx.x * 4 + wave;
    const int tileBase = tile * 16;          // flat position index base
    const int lo   = lane >> 4;              // lane half: 0 or 1
    const int mrow = lane & 15;              // row (A) or column (B/C) index

    const int pos = tileBase + mrow;
    const float* arow = input + (size_t)pos * DD;
    const int tgt = target[pos];
    const float* trow = emb_weight + (size_t)tgt * DD;

    v8f acc[NTILES];
    v8f accT;
#pragma unroll
    for (int t = 0; t < NTILES; ++t)
#pragma unroll
        for (int i = 0; i < 8; ++i) acc[t][i] = 0.0f;
#pragma unroll
    for (int i = 0; i < 8; ++i) accT[i] = 0.0f;

    for (int d0 = 0; d0 < DD; d0 += 32) {
        // ---- A tile (16x32 bf16): lane holds row mrow.
        // halves 0..7  -> K = d0 + 16*lo + 0..7
        // halves 8..15 -> K = d0 + 16*lo + 8 + 0..7
        v16bf a;
        {
            const float4* p0 = (const float4*)(arow + d0 + 16 * lo);
            const float4* p1 = (const float4*)(arow + d0 + 16 * lo + 8);
            float4 f0 = p0[0], f1 = p0[1];
            float4 g0 = p1[0], g1 = p1[1];
            a[0] = (__bf16)f0.x; a[1] = (__bf16)f0.y; a[2] = (__bf16)f0.z; a[3] = (__bf16)f0.w;
            a[4] = (__bf16)f1.x; a[5] = (__bf16)f1.y; a[6] = (__bf16)f1.z; a[7] = (__bf16)f1.w;
            a[8]  = (__bf16)g0.x; a[9]  = (__bf16)g0.y; a[10] = (__bf16)g0.z; a[11] = (__bf16)g0.w;
            a[12] = (__bf16)g1.x; a[13] = (__bf16)g1.y; a[14] = (__bf16)g1.z; a[15] = (__bf16)g1.w;
        }

        // ---- Target B tile (32x16 bf16): lane holds column mrow,
        // K rows d0 + 16*lo + 0..15 (contiguous floats of the gathered row).
        v16bf bt;
        {
            const float4* p = (const float4*)(trow + d0 + 16 * lo);
            float4 f0 = p[0], f1 = p[1], f2 = p[2], f3 = p[3];
            bt[0]  = (__bf16)f0.x; bt[1]  = (__bf16)f0.y; bt[2]  = (__bf16)f0.z; bt[3]  = (__bf16)f0.w;
            bt[4]  = (__bf16)f1.x; bt[5]  = (__bf16)f1.y; bt[6]  = (__bf16)f1.z; bt[7]  = (__bf16)f1.w;
            bt[8]  = (__bf16)f2.x; bt[9]  = (__bf16)f2.y; bt[10] = (__bf16)f2.z; bt[11] = (__bf16)f2.w;
            bt[12] = (__bf16)f3.x; bt[13] = (__bf16)f3.y; bt[14] = (__bf16)f3.z; bt[15] = (__bf16)f3.w;
        }

        // ---- Noise tiles: bf16 loaded straight from workspace (L1/L2 hot).
#pragma unroll
        for (int t = 0; t < NTILES; ++t) {
            const v16bf* bp = (const v16bf*)(noise_bf16 +
                ((size_t)(t * 16 + mrow) * DD + d0 + 16 * lo));
            v16bf bn = *bp;
            acc[t] = __builtin_amdgcn_wmma_f32_16x16x32_bf16(
                false, a, false, bn, (short)0, acc[t], false, false);
        }
        accT = __builtin_amdgcn_wmma_f32_16x16x32_bf16(
            false, a, false, bt, (short)0, accT, false, false);
    }

    // ---- Epilogue: C element (vgpr r, lane) = [M = r + 8*lo, N = mrow]
    float sum = 0.0f;
#pragma unroll
    for (int t = 0; t < NTILES; ++t) {
        const int col = t * 16 + mrow;          // noise column (same for all r)
        const float cn = c_n[col];
        const bool valid = (col < KK);
#pragma unroll
        for (int r = 0; r < 8; ++r) {
            float x = acc[t][r] + cn;           // x_n
            sum += valid ? -softplus_f(x) : 0.0f;   // log_sigmoid(-x_n)
        }
    }

    // Target diagonal: lives where N == M, i.e. (mrow >> 3) == lo, r = mrow&7.
    if ((mrow >> 3) == lo) {
        const float NORM = logf((float)VV);
        const float LOGK = logf((float)KK);
        const int rd = mrow & 7;
        float xt = accT[rd] + emb_bias[tgt] - NORM - logprob_noise[tgt] - LOGK;
        sum += -softplus_f(-xt);                // log_sigmoid(x_t)
    }

    // Wave32 reduction; one atomic per wave. All 16 positions of this tile
    // share one batch row: b = tileBase / N.
#pragma unroll
    for (int off = 16; off > 0; off >>= 1)
        sum += __shfl_xor(sum, off, 32);
    if (lane == 0)
        atomicAdd(out + (tileBase >> 8), sum);
}

extern "C" void kernel_launch(void* const* d_in, const int* in_sizes, int n_in,
                              void* d_out, int out_size, void* d_ws, size_t ws_size,
                              hipStream_t stream) {
    const float* input         = (const float*)d_in[0];
    const float* emb_weight    = (const float*)d_in[1];
    const float* emb_bias      = (const float*)d_in[2];
    const float* logprob_noise = (const float*)d_in[3];
    const int*   target        = (const int*)d_in[4];
    const int*   noise_samples = (const int*)d_in[5];
    float* out = (float*)d_out;

    unsigned short* noise_bf16 = (unsigned short*)d_ws;
    float* c_n = (float*)((char*)d_ws + (size_t)KPAD * DD * sizeof(unsigned short));

    zero_out_kernel<<<1, BB, 0, stream>>>(out);
    prep_noise_kernel<<<KPAD, 128, 0, stream>>>(emb_weight, emb_bias,
                                                logprob_noise, noise_samples,
                                                noise_bf16, c_n);
    const int totalTiles = (BB * NN) / 16;   // 1024 waves of work
    nce_main_kernel<<<totalTiles / 4, 128, 0, stream>>>(
        input, emb_weight, emb_bias, logprob_noise, target,
        noise_bf16, c_n, out);
}
